// MultiHeadSelfAttention_61418032333107
// MI455X (gfx1250) — matrix operand order repaired and checked by reading, not compiled, not run on hardware
//
#include <hip/hip_runtime.h>
#include <math.h>
#include <stdint.h>

typedef _Float16 half_t;
typedef __attribute__((ext_vector_type(16))) _Float16 v16h;
typedef __attribute__((ext_vector_type(8)))  _Float16 v8h;
typedef __attribute__((ext_vector_type(8)))  float    v8f;
typedef __attribute__((ext_vector_type(4)))  int      v4i;

#define EMB   768
#define SEQ   1024
#define BATCH 16
#define HEADS 12
#define HDIM  64
#define MROWS (BATCH * SEQ)   // 16384

#ifndef __has_builtin
#define __has_builtin(x) 0
#endif
#if __has_builtin(__builtin_amdgcn_global_load_async_to_lds_b128) && \
    __has_builtin(__builtin_amdgcn_s_wait_asynccnt)
#define USE_ASYNC 1
#else
#define USE_ASYNC 0
#endif

#if USE_ASYNC
#define ASYNC_WAIT(n) __builtin_amdgcn_s_wait_asynccnt(n)
typedef __attribute__((address_space(1))) v4i* gv4_ptr;   // global (AS1)
typedef __attribute__((address_space(3))) v4i* lv4_ptr;   // LDS (AS3)
#else
#define ASYNC_WAIT(n)
#endif

// copy 16 bytes global -> LDS (async DMA engine if available)
__device__ __forceinline__ void cp16(const half_t* g, half_t* l) {
#if USE_ASYNC
  // AS1 ptr == 64-bit VA; AS3 ptr == low 32 bits of the generic LDS address
  // so integer round-trips are exact.
  gv4_ptr gp = (gv4_ptr)(uintptr_t)g;
  lv4_ptr lp = (lv4_ptr)(uint32_t)(uintptr_t)l;
  __builtin_amdgcn_global_load_async_to_lds_b128(gp, lp, 0, 0);
#else
  *(v8h*)l = *(const v8h*)g;
#endif
}

union AFrag { v16h v; v8h c[2]; };

__device__ __forceinline__ v8f wmma16(v16h a, v16h b, v8f c) {
  // D = A(16x32 f16) * B(32x16 f16) + C(16x16 f32)
  return __builtin_amdgcn_wmma_f32_16x16x32_f16(false, a, false, b, (short)0, c,
                                                false, false);
}

// A-fragment: lane holds row = lane&15; halfs e0..7 -> K = kb+0..7,
// e8..15 -> K = kb+16..23, kb = (lane>=16)?8:0. Two 16B chunks.
__device__ __forceinline__ v16h frag_a(const half_t* base, int stride, int lane) {
  int row = lane & 15;
  int kb  = (lane >> 4) << 3;
  AFrag f;
  f.c[0] = *(const v8h*)(base + row * stride + kb);
  f.c[1] = *(const v8h*)(base + row * stride + kb + 16);
  return f.v;
}

// B-fragment from B^T storage ([N][K] row-major): lane holds col n = lane&15;
// halfs e0..7 -> K = ko+0..7, e8..15 -> K = ko+16..23, ko = (lane>=16)?8:0.
__device__ __forceinline__ v16h frag_b(const half_t* base, int stride, int lane) {
  int n  = lane & 15;
  int ko = (lane >> 4) << 3;
  AFrag f;
  f.c[0] = *(const v8h*)(base + n * stride + ko);
  f.c[1] = *(const v8h*)(base + n * stride + ko + 16);
  return f.v;
}

// ---------------- conversion kernels ----------------

__global__ __launch_bounds__(256) void cvt_f32_to_f16(const float* __restrict__ in,
                                                      half_t* __restrict__ out, int n) {
  int i = (blockIdx.x * 256 + threadIdx.x) * 4;
  if (i + 3 < n) {
#pragma unroll
    for (int t = 0; t < 4; ++t) out[i + t] = (half_t)in[i + t];
  }
}

// W[K][N] f32 -> Wt[N][K] f16
__global__ __launch_bounds__(256) void cvt_transpose(const float* __restrict__ W,
                                                     half_t* __restrict__ Wt,
                                                     int K, int N) {
  int idx = blockIdx.x * 256 + threadIdx.x;
  if (idx < K * N) {
    int k = idx / N, n = idx - k * N;
    Wt[(size_t)n * K + k] = (half_t)W[idx];
  }
}

// ---------------- generic f16 WMMA GEMM ----------------
// C[M x N] = A[M x K] * Bt[N x K]^T + bias, M tiles of 128.
// mode 0: QKV epilogue (scatter into Q (scaled), K, Vt layouts)
// mode 1: f32 output (projection)
// Double-buffered LDS; next K-tile staged via async global->LDS DMA while the
// current tile feeds the WMMA pipes; s_wait_asynccnt(4) leaves only the four
// just-issued per-wave async ops outstanding.

#define GT_M 128
#define GT_N 128
#define GT_K 32
#define GT_LD 40   // padded LDS stride in halfs (80 B, 16B multiple)

__global__ __launch_bounds__(256) void gemm_f16(
    const half_t* __restrict__ A, const half_t* __restrict__ Bt,
    const float* __restrict__ bias, int N, int Kdim, int mode,
    half_t* __restrict__ Qb, half_t* __restrict__ Kb, half_t* __restrict__ Vt,
    float* __restrict__ Out) {
  __shared__ half_t As[2][GT_M * GT_LD];
  __shared__ half_t Bs[2][GT_N * GT_LD];

  int tid  = threadIdx.x;
  int lane = tid & 31;
  int w    = tid >> 5;       // 8 waves
  int wm   = w & 3;          // 4 waves along M
  int wn   = w >> 2;         // 2 waves along N
  int m0   = blockIdx.x * GT_M;
  int n0   = blockIdx.y * GT_N;

  v8f acc[2][4] = {};

  int ar = tid >> 1;               // 0..127
  int ac = (tid & 1) * 16;         // 0 or 16 halfs

  auto issue_tile = [&](int k0, int buf) {
    const half_t* ga = A  + (size_t)(m0 + ar) * Kdim + k0 + ac;
    const half_t* gb = Bt + (size_t)(n0 + ar) * Kdim + k0 + ac;
    cp16(ga,     &As[buf][ar * GT_LD + ac]);
    cp16(ga + 8, &As[buf][ar * GT_LD + ac + 8]);
    cp16(gb,     &Bs[buf][ar * GT_LD + ac]);
    cp16(gb + 8, &Bs[buf][ar * GT_LD + ac + 8]);
  };

  int nk = Kdim / GT_K;
  issue_tile(0, 0);
  for (int it = 0; it < nk; ++it) {
    int cur = it & 1;
    if (it + 1 < nk) {
      issue_tile((it + 1) * GT_K, cur ^ 1);
      ASYNC_WAIT(4);   // previous tile landed; next tile still in flight
    } else {
      ASYNC_WAIT(0);
    }
    __syncthreads();

    v16h af[2], bf[4];
#pragma unroll
    for (int i = 0; i < 2; ++i)
      af[i] = frag_a(&As[cur][(wm * 32 + i * 16) * GT_LD], GT_LD, lane);
#pragma unroll
    for (int j = 0; j < 4; ++j)
      bf[j] = frag_b(&Bs[cur][(wn * 64 + j * 16) * GT_LD], GT_LD, lane);
#pragma unroll
    for (int i = 0; i < 2; ++i)
#pragma unroll
      for (int j = 0; j < 4; ++j)
        acc[i][j] = wmma16(af[i], bf[j], acc[i][j]);
    __syncthreads();   // readers done before this buffer is overwritten
  }

  // epilogue: C element (r) maps to row = r + 8*(lane>=16), col = lane&15
  int hi = (lane >> 4) << 3;
  int nl = lane & 15;
#pragma unroll
  for (int i = 0; i < 2; ++i) {
#pragma unroll
    for (int j = 0; j < 4; ++j) {
#pragma unroll
      for (int r = 0; r < 8; ++r) {
        int m = m0 + wm * 32 + i * 16 + hi + r;
        int n = n0 + wn * 64 + j * 16 + nl;
        float v = acc[i][j][r] + bias[n];
        if (mode == 0) {
          int which = (n < EMB) ? 0 : ((n < 2 * EMB) ? 1 : 2);
          int e = n - which * EMB;
          int h = e >> 6, d = e & 63;
          int b = m >> 10, s = m & 1023;
          size_t bh = (size_t)(b * HEADS + h);
          if (which == 0)      Qb[(bh * SEQ + s) * HDIM + d] = (half_t)(v * 0.125f);
          else if (which == 1) Kb[(bh * SEQ + s) * HDIM + d] = (half_t)v;
          else                 Vt[(bh * HDIM + d) * SEQ + s] = (half_t)v;
        } else {
          Out[(size_t)m * N + n] = v;
        }
      }
    }
  }
}

// ---------------- flash attention ----------------
// grid: (S/128, B*H). 8 waves; wave w handles 16 query rows.
// Q pre-scaled by 1/sqrt(D). K [bh][S][D], Vt [bh][D][S], out Ah [B][S][E] f16.
// K/V tiles double-buffered through the async global->LDS engine.

#define AT_LD 72   // padded stride (144 B)

__global__ __launch_bounds__(256) void attn_flash(
    const half_t* __restrict__ Q, const half_t* __restrict__ K,
    const half_t* __restrict__ Vt, half_t* __restrict__ Ah) {
  __shared__ half_t Ks[2][64 * AT_LD];
  __shared__ half_t Vs[2][64 * AT_LD];
  __shared__ half_t Ps[128 * AT_LD];

  int tid  = threadIdx.x;
  int lane = tid & 31;
  int w    = tid >> 5;
  int qt   = blockIdx.x;           // 0..7
  int bh   = blockIdx.y;           // 0..191
  int b    = bh / HEADS;
  int h    = bh - b * HEADS;

  const half_t* Qg = Q  + (size_t)bh * SEQ * HDIM;
  const half_t* Kg = K  + (size_t)bh * SEQ * HDIM;
  const half_t* Vg = Vt + (size_t)bh * HDIM * SEQ;

  int qrow0 = qt * 128 + w * 16;

  // load Q fragments (2 k-steps over D=64) straight from global
  v16h qf[2];
  {
    int row = qrow0 + (lane & 15);
    int kb  = (lane >> 4) << 3;
#pragma unroll
    for (int f = 0; f < 2; ++f) {
      AFrag a;
      a.c[0] = *(const v8h*)&Qg[row * HDIM + f * 32 + kb];
      a.c[1] = *(const v8h*)&Qg[row * HDIM + f * 32 + kb + 16];
      qf[f] = a.v;
    }
  }

  float mrow[8], lrow[8];
#pragma unroll
  for (int r = 0; r < 8; ++r) { mrow[r] = -1e30f; lrow[r] = 0.0f; }
  v8f oacc[4] = {};

  int hi = (lane >> 4) << 3;
  int rr = tid >> 2;               // 0..63
  int cc = (tid & 3) * 16;         // 0,16,32,48

  auto issue_kv = [&](int kt, int buf) {
    int key0 = kt * 64;
    cp16(&Kg[(key0 + rr) * HDIM + cc],     &Ks[buf][rr * AT_LD + cc]);
    cp16(&Kg[(key0 + rr) * HDIM + cc + 8], &Ks[buf][rr * AT_LD + cc + 8]);
    cp16(&Vg[rr * SEQ + key0 + cc],        &Vs[buf][rr * AT_LD + cc]);
    cp16(&Vg[rr * SEQ + key0 + cc + 8],    &Vs[buf][rr * AT_LD + cc + 8]);
  };

  const int nkt = SEQ / 64;
  issue_kv(0, 0);
  for (int kt = 0; kt < nkt; ++kt) {
    int cur = kt & 1;
    if (kt + 1 < nkt) {
      issue_kv(kt + 1, cur ^ 1);
      ASYNC_WAIT(4);
    } else {
      ASYNC_WAIT(0);
    }
    __syncthreads();

    // scores: S = Q K^T (16 q rows x 64 keys per wave)
    v8f sacc[4] = {};
#pragma unroll
    for (int f = 0; f < 2; ++f) {
#pragma unroll
      for (int j = 0; j < 4; ++j) {
        v16h bf = frag_b(&Ks[cur][(j * 16) * AT_LD + f * 32], AT_LD, lane);
        sacc[j] = wmma16(qf[f], bf, sacc[j]);
      }
    }

    // online softmax (rows split 16-lane halves; xor 1/2/4/8 stays in-half)
#pragma unroll
    for (int r = 0; r < 8; ++r) {
      float mx = fmaxf(fmaxf(sacc[0][r], sacc[1][r]),
                       fmaxf(sacc[2][r], sacc[3][r]));
#pragma unroll
      for (int d = 8; d >= 1; d >>= 1) mx = fmaxf(mx, __shfl_xor(mx, d, 32));
      float mnew  = fmaxf(mrow[r], mx);
      float alpha = __expf(mrow[r] - mnew);
      mrow[r] = mnew;
      float rs = 0.0f;
#pragma unroll
      for (int j = 0; j < 4; ++j) {
        float p = __expf(sacc[j][r] - mnew);
        sacc[j][r] = p;
        rs += p;
      }
#pragma unroll
      for (int d = 8; d >= 1; d >>= 1) rs += __shfl_xor(rs, d, 32);
      lrow[r] = lrow[r] * alpha + rs;
#pragma unroll
      for (int j = 0; j < 4; ++j) oacc[j][r] *= alpha;
    }

    // stage P (C-layout -> LDS -> A-layout); wave-private region, DS in-order
#pragma unroll
    for (int j = 0; j < 4; ++j)
#pragma unroll
      for (int r = 0; r < 8; ++r)
        Ps[(w * 16 + hi + r) * AT_LD + j * 16 + (lane & 15)] = (half_t)sacc[j][r];

    // O += P V  (k-dim = 64 keys -> 2 k-steps)
#pragma unroll
    for (int f = 0; f < 2; ++f) {
      v16h pf = frag_a(&Ps[(w * 16) * AT_LD + f * 32], AT_LD, lane);
#pragma unroll
      for (int j = 0; j < 4; ++j) {
        v16h vf = frag_b(&Vs[cur][(j * 16) * AT_LD + f * 32], AT_LD, lane);
        oacc[j] = wmma16(pf, vf, oacc[j]);
      }
    }
    __syncthreads();
  }

  // normalize and write [B][S][E] f16
  int nl = lane & 15;
#pragma unroll
  for (int r = 0; r < 8; ++r) {
    float inv = 1.0f / lrow[r];
    int s = qrow0 + hi + r;
#pragma unroll
    for (int j = 0; j < 4; ++j) {
      int col = h * HDIM + j * 16 + nl;
      Ah[((size_t)b * SEQ + s) * EMB + col] = (half_t)(oacc[j][r] * inv);
    }
  }
}

// ---------------- host launch ----------------

extern "C" void kernel_launch(void* const* d_in, const int* in_sizes, int n_in,
                              void* d_out, int out_size, void* d_ws, size_t ws_size,
                              hipStream_t stream) {
  (void)in_sizes; (void)n_in; (void)out_size; (void)ws_size;
  const float* x      = (const float*)d_in[0];   // [16,1024,768]
  const float* qkv_w  = (const float*)d_in[1];   // [768,2304]
  const float* qkv_b  = (const float*)d_in[2];   // [2304]
  const float* proj_w = (const float*)d_in[3];   // [768,768]
  const float* proj_b = (const float*)d_in[4];   // [768]
  float* out = (float*)d_out;                    // [16,1024,768]

  const size_t N_X  = (size_t)MROWS * EMB;            // 12,582,912
  const size_t N_WQ = (size_t)EMB * 3 * EMB;          // 1,769,472
  const size_t N_WP = (size_t)EMB * EMB;              // 589,824
  const size_t N_H  = (size_t)BATCH * HEADS * SEQ * HDIM; // 12,582,912

  half_t* Xh  = (half_t*)d_ws;
  half_t* WqT = Xh  + N_X;
  half_t* WpT = WqT + N_WQ;
  half_t* Qb  = WpT + N_WP;
  half_t* Kb  = Qb  + N_H;
  half_t* Vt  = Kb  + N_H;
  half_t* Ah  = Vt  + N_H;

  // 1) precision conversion + weight transposes
  cvt_f32_to_f16<<<dim3((N_X / 4 + 255) / 256), 256, 0, stream>>>(x, Xh, (int)N_X);
  cvt_transpose<<<dim3((N_WQ + 255) / 256), 256, 0, stream>>>(qkv_w, WqT, EMB, 3 * EMB);
  cvt_transpose<<<dim3((N_WP + 255) / 256), 256, 0, stream>>>(proj_w, WpT, EMB, EMB);

  // 2) fused QKV projection -> Q (scaled), K, V^T head layouts
  gemm_f16<<<dim3(MROWS / GT_M, (3 * EMB) / GT_N), 256, 0, stream>>>(
      Xh, WqT, qkv_b, 3 * EMB, EMB, 0, Qb, Kb, Vt, nullptr);

  // 3) flash attention per (b,h,q-tile)
  attn_flash<<<dim3(SEQ / 128, BATCH * HEADS), 256, 0, stream>>>(Qb, Kb, Vt, Ah);

  // 4) output projection -> f32
  gemm_f16<<<dim3(MROWS / GT_M, EMB / GT_N), 256, 0, stream>>>(
      Ah, WpT, proj_b, EMB, EMB, 1, nullptr, nullptr, nullptr, out);
}
